// Block_6519760355385
// MI455X (gfx1250) — matrix operand order repaired and checked by reading, not compiled, not run on hardware
//
#include <hip/hip_runtime.h>
#include <hip/hip_bf16.h>

// ---------------------------------------------------------------------------
// Transformer block for MI455X (gfx1250): every matmul runs on
// v_wmma_f32_16x16x32_bf16 (wave32). FP32 model weights/activations are
// converted to bf16 on device; accumulation stays in f32.
// GEMM wave tile is 32x64 (2x4 WMMA tiles) to maximize WMMA issue density
// vs VMEM issue (12 b128 loads per 8 WMMA per k-step).
// ---------------------------------------------------------------------------

typedef __attribute__((ext_vector_type(16))) __bf16 v16bf;
typedef __attribute__((ext_vector_type(8)))  float  v8f;

#define D_    768
#define H_    12
#define HD_   64
#define DFF_  3072

#define MODE_F32_RES   0
#define MODE_BF16_RELU 1
#define MODE_QK        2
#define MODE_VT        3

struct U16x16 { uint4 a, b; };

__device__ __forceinline__ unsigned short f2bfu(float f) {
  unsigned u = __builtin_bit_cast(unsigned, f);
  return (unsigned short)((u + 0x7FFFu + ((u >> 16) & 1u)) >> 16);
}

__device__ __forceinline__ v16bf make_frag(uint4 c0, uint4 c1) {
  U16x16 t; t.a = c0; t.b = c1;
  return __builtin_bit_cast(v16bf, t);
}

// A-matrix fragment (16x32 bf16, MxK). ISA layout: lane L (hf=L>>4, m=L&15)
// holds row m; elements e: K = (e>>3)*16 + hf*8 + (e&7)
// -> two contiguous 8-element chunks at k0+hf*8 and k0+16+hf*8.
__device__ __forceinline__ v16bf load_a_frag(const unsigned short* A, int lda,
                                             int row0, int k0, int lane) {
  int hf = lane >> 4, m = lane & 15;
  const unsigned short* p = A + (size_t)(row0 + m) * lda + (k0 + hf * 8);
  return make_frag(*(const uint4*)p, *(const uint4*)(p + 16));
}

// B-matrix fragment (32x16 bf16, KxN) from a transposed [N,K] array.
// ISA layout: lane L holds column n=L&15; elements e: K = (e>>3)*16 + hf*8 + (e&7)
// -> two contiguous 8-element chunks at k0+hf*8 and k0+16+hf*8.
__device__ __forceinline__ v16bf load_b_frag(const unsigned short* Bt, int ldb,
                                             int col0, int k0, int lane) {
  int hf = lane >> 4, n = lane & 15;
  const unsigned short* p = Bt + (size_t)(col0 + n) * ldb + (k0 + hf * 8);
  return make_frag(*(const uint4*)p, *(const uint4*)(p + 16));
}

// ---------------------------------------------------------------------------
// Weight convert + transpose: fp32 [K,N] row-major -> bf16 [N,K] row-major
// ---------------------------------------------------------------------------
__global__ void convt_kernel(const float* __restrict__ W,
                             unsigned short* __restrict__ Wt, int K, int N) {
  int idx = blockIdx.x * 256 + threadIdx.x;
  if (idx >= K * N) return;
  int k = idx / N, n = idx % N;
  Wt[(size_t)n * K + k] = f2bfu(W[idx]);
}

// ---------------------------------------------------------------------------
// LayerNorm (row = one workgroup) -> bf16 output row
// ---------------------------------------------------------------------------
__global__ __launch_bounds__(256) void ln_bf16_kernel(
    const float* __restrict__ x, const float* __restrict__ g,
    const float* __restrict__ b, unsigned short* __restrict__ out) {
  __shared__ float r1[256], r2[256];
  int row = blockIdx.x, tid = threadIdx.x;
  const float* xr = x + (size_t)row * D_;
  float s1 = 0.f, s2 = 0.f;
  for (int c = tid; c < D_; c += 256) { float v = xr[c]; s1 += v; s2 += v * v; }
  r1[tid] = s1; r2[tid] = s2; __syncthreads();
  for (int o = 128; o > 0; o >>= 1) {
    if (tid < o) { r1[tid] += r1[tid + o]; r2[tid] += r2[tid + o]; }
    __syncthreads();
  }
  float mean = r1[0] * (1.f / D_);
  float var  = r2[0] * (1.f / D_) - mean * mean;
  float rs   = rsqrtf(var + 1e-5f);
  for (int c = tid; c < D_; c += 256)
    out[(size_t)row * D_ + c] = f2bfu((xr[c] - mean) * rs * g[c] + b[c]);
}

// ---------------------------------------------------------------------------
// Generic WMMA GEMM: C[M,N] = A[M,K](bf16) @ Bt[N,K](bf16)^T + bias
// Block tile 128x128, 8 waves (4Mx2N), wave tile 32x64 = 2x4 WMMA tiles.
// Per k-step per wave: 12 global_load_b128 feed 8 v_wmma.
// ---------------------------------------------------------------------------
__global__ __launch_bounds__(256) void gemm_bf16_kernel(
    const unsigned short* __restrict__ A, const unsigned short* __restrict__ Bt,
    const float* __restrict__ bias, const float* __restrict__ res,
    float* __restrict__ outF, unsigned short* __restrict__ outB,
    int M, int N, int K, int S_, int mode) {
  int lane = threadIdx.x & 31;
  int wid  = threadIdx.x >> 5;
  int rowBase = blockIdx.y * 128 + (wid & 3) * 32;
  int colBase = blockIdx.x * 128 + (wid >> 2) * 64;

  v8f acc[2][4] = {};
  for (int k = 0; k < K; k += 32) {
    v16bf a0 = load_a_frag(A, K, rowBase,      k, lane);
    v16bf a1 = load_a_frag(A, K, rowBase + 16, k, lane);
    v16bf b0 = load_b_frag(Bt, K, colBase,      k, lane);
    v16bf b1 = load_b_frag(Bt, K, colBase + 16, k, lane);
    v16bf b2 = load_b_frag(Bt, K, colBase + 32, k, lane);
    v16bf b3 = load_b_frag(Bt, K, colBase + 48, k, lane);
    acc[0][0] = __builtin_amdgcn_wmma_f32_16x16x32_bf16(false, a0, false, b0, (short)0, acc[0][0], false, false);
    acc[0][1] = __builtin_amdgcn_wmma_f32_16x16x32_bf16(false, a0, false, b1, (short)0, acc[0][1], false, false);
    acc[0][2] = __builtin_amdgcn_wmma_f32_16x16x32_bf16(false, a0, false, b2, (short)0, acc[0][2], false, false);
    acc[0][3] = __builtin_amdgcn_wmma_f32_16x16x32_bf16(false, a0, false, b3, (short)0, acc[0][3], false, false);
    acc[1][0] = __builtin_amdgcn_wmma_f32_16x16x32_bf16(false, a1, false, b0, (short)0, acc[1][0], false, false);
    acc[1][1] = __builtin_amdgcn_wmma_f32_16x16x32_bf16(false, a1, false, b1, (short)0, acc[1][1], false, false);
    acc[1][2] = __builtin_amdgcn_wmma_f32_16x16x32_bf16(false, a1, false, b2, (short)0, acc[1][2], false, false);
    acc[1][3] = __builtin_amdgcn_wmma_f32_16x16x32_bf16(false, a1, false, b3, (short)0, acc[1][3], false, false);
  }

  // C layout: VGPR r, lane L -> m = tile_row + (L>>4)*8 + r, n = tile_col + (L&15)
  int hf = lane >> 4, n15 = lane & 15;
#pragma unroll
  for (int tm = 0; tm < 2; ++tm)
#pragma unroll
    for (int tn = 0; tn < 4; ++tn)
#pragma unroll
      for (int r = 0; r < 8; ++r) {
        int m = rowBase + tm * 16 + hf * 8 + r;
        int n = colBase + tn * 16 + n15;
        float v = acc[tm][tn][r] + bias[n];
        if (mode == MODE_F32_RES) {
          outF[(size_t)m * N + n] = v + res[(size_t)m * N + n];
        } else if (mode == MODE_BF16_RELU) {
          outB[(size_t)m * N + n] = f2bfu(fmaxf(v, 0.f));
        } else {
          int bb = m / S_, s = m % S_;
          int hh = n >> 6, hd = n & 63;
          size_t idx;
          if (mode == MODE_QK)
            idx = (((size_t)bb * H_ + hh) * S_ + s) * HD_ + hd;       // [B,H,S,64]
          else
            idx = (((size_t)bb * H_ + hh) * HD_ + hd) * (size_t)S_ + s; // [B,H,64,S]
          outB[idx] = f2bfu(v);
        }
      }
}

// ---------------------------------------------------------------------------
// Flash attention: 1 wave per (batch*head, 16-query tile). Causal + mask.
// QK^T and P@V on WMMA; P staged via LDS into A-fragment layout.
// ---------------------------------------------------------------------------
__global__ __launch_bounds__(32) void attn_kernel(
    const unsigned short* __restrict__ Qh, const unsigned short* __restrict__ Kh,
    const unsigned short* __restrict__ Vt, const int* __restrict__ amask,
    unsigned short* __restrict__ ctxB, int S_) {
  __shared__ unsigned short Plds[16 * 32];
  int lane = threadIdx.x & 31;
  int hf = lane >> 4, n15 = lane & 15;
  int q0 = blockIdx.x * 16;
  int bh = blockIdx.y;
  int b = bh / H_, h = bh % H_;
  const unsigned short* Qb = Qh + (size_t)bh * S_ * HD_;
  const unsigned short* Kb = Kh + (size_t)bh * S_ * HD_;
  const unsigned short* Vb = Vt + (size_t)bh * HD_ * S_;
  const int* mk = amask + (size_t)b * S_;

  // Q fragments invariant over key blocks (contraction d: k-steps 0 and 32)
  v16bf qa0 = load_a_frag(Qb, HD_, q0, 0,  lane);
  v16bf qa1 = load_a_frag(Qb, HD_, q0, 32, lane);

  v8f acc[4] = {};
  float mi[8], li[8];
#pragma unroll
  for (int r = 0; r < 8; ++r) { mi[r] = -3.0e38f; li[r] = 0.f; }

  int nkb = (q0 + 16 + 31) >> 5;  // key blocks of 32 covering causal range
  for (int kb = 0; kb < nkb; ++kb) {
    int key0 = kb * 32;
    v8f s[2] = {};
#pragma unroll
    for (int st = 0; st < 2; ++st) {
      v16bf kf0 = load_b_frag(Kb, HD_, key0 + st * 16, 0,  lane);
      v16bf kf1 = load_b_frag(Kb, HD_, key0 + st * 16, 32, lane);
      s[st] = __builtin_amdgcn_wmma_f32_16x16x32_bf16(false, qa0, false, kf0, (short)0, s[st], false, false);
      s[st] = __builtin_amdgcn_wmma_f32_16x16x32_bf16(false, qa1, false, kf1, (short)0, s[st], false, false);
    }
    // scale + causal/attention mask
#pragma unroll
    for (int st = 0; st < 2; ++st) {
      int key = key0 + st * 16 + n15;
      int mval = mk[key];
#pragma unroll
      for (int r = 0; r < 8; ++r) {
        int qrow = q0 + hf * 8 + r;
        float v = s[st][r] * 0.125f;  // 1/sqrt(64)
        s[st][r] = (key > qrow || mval == 0) ? -3.0e38f : v;
      }
    }
    // online softmax (row r+8*hf lives across the 16 lanes of a half)
#pragma unroll
    for (int r = 0; r < 8; ++r) {
      float mx = fmaxf(s[0][r], s[1][r]);
#pragma unroll
      for (int o = 8; o >= 1; o >>= 1) mx = fmaxf(mx, __shfl_xor(mx, o, 32));
      float mnew = fmaxf(mi[r], mx);
      float fac = __expf(mi[r] - mnew);
      float p0 = __expf(s[0][r] - mnew);
      float p1 = __expf(s[1][r] - mnew);
      s[0][r] = p0; s[1][r] = p1;
      float rsum = p0 + p1;
#pragma unroll
      for (int o = 8; o >= 1; o >>= 1) rsum += __shfl_xor(rsum, o, 32);
      li[r] = li[r] * fac + rsum;
      mi[r] = mnew;
#pragma unroll
      for (int t = 0; t < 4; ++t) acc[t][r] *= fac;
    }
    // stage P (16x32 bf16) through LDS into A-fragment layout
    __syncthreads();
#pragma unroll
    for (int st = 0; st < 2; ++st)
#pragma unroll
      for (int r = 0; r < 8; ++r)
        Plds[(hf * 8 + r) * 32 + st * 16 + n15] = f2bfu(s[st][r]);
    __syncthreads();
    const unsigned short* pp = Plds + n15 * 32 + hf * 8;
    v16bf pa = make_frag(*(const uint4*)pp, *(const uint4*)(pp + 16));
#pragma unroll
    for (int t = 0; t < 4; ++t) {
      v16bf vf = load_b_frag(Vb, S_, t * 16, key0, lane);  // V^T: rows=d, cols=key
      acc[t] = __builtin_amdgcn_wmma_f32_16x16x32_bf16(false, pa, false, vf, (short)0, acc[t], false, false);
    }
  }
  // normalize and write ctx as bf16 [B,S,D] rows (head-interleaved cols)
#pragma unroll
  for (int r = 0; r < 8; ++r) {
    float inv = 1.f / li[r];
    int q = q0 + hf * 8 + r;
#pragma unroll
    for (int t = 0; t < 4; ++t)
      ctxB[(size_t)(b * S_ + q) * D_ + h * HD_ + t * 16 + n15] = f2bfu(acc[t][r] * inv);
  }
}

// ---------------------------------------------------------------------------
extern "C" void kernel_launch(void* const* d_in, const int* in_sizes, int n_in,
                              void* d_out, int out_size, void* d_ws, size_t ws_size,
                              hipStream_t stream) {
  const float* x      = (const float*)d_in[0];
  const int*   amask  = (const int*)  d_in[1];
  const float* ln1_g  = (const float*)d_in[2];
  const float* ln1_b  = (const float*)d_in[3];
  const float* ln2_g  = (const float*)d_in[4];
  const float* ln2_b  = (const float*)d_in[5];
  const float* Wq = (const float*)d_in[6];  const float* bq = (const float*)d_in[7];
  const float* Wk = (const float*)d_in[8];  const float* bk = (const float*)d_in[9];
  const float* Wv = (const float*)d_in[10]; const float* bv = (const float*)d_in[11];
  const float* Wo = (const float*)d_in[12]; const float* bo = (const float*)d_in[13];
  const float* W1 = (const float*)d_in[14]; const float* b1 = (const float*)d_in[15];
  const float* W2 = (const float*)d_in[16]; const float* b2 = (const float*)d_in[17];

  const int B = 2, S = 4096, MS = B * S;

  char* ws = (char*)d_ws;
  size_t off = 0;
  auto alloc = [&](size_t bytes) -> void* {
    void* p = ws + off;
    off = (off + bytes + 255) & ~(size_t)255;
    return p;
  };
  unsigned short* WqT  = (unsigned short*)alloc((size_t)D_ * D_ * 2);
  unsigned short* WkT  = (unsigned short*)alloc((size_t)D_ * D_ * 2);
  unsigned short* WvT  = (unsigned short*)alloc((size_t)D_ * D_ * 2);
  unsigned short* WoT  = (unsigned short*)alloc((size_t)D_ * D_ * 2);
  unsigned short* W1T  = (unsigned short*)alloc((size_t)D_ * DFF_ * 2);
  unsigned short* W2T  = (unsigned short*)alloc((size_t)D_ * DFF_ * 2);
  unsigned short* hB   = (unsigned short*)alloc((size_t)MS * D_ * 2);
  unsigned short* Qh   = (unsigned short*)alloc((size_t)MS * D_ * 2);
  unsigned short* Kh   = (unsigned short*)alloc((size_t)MS * D_ * 2);
  unsigned short* Vt   = (unsigned short*)alloc((size_t)MS * D_ * 2);
  unsigned short* ctxB = (unsigned short*)alloc((size_t)MS * D_ * 2);
  float*          x2   = (float*)         alloc((size_t)MS * D_ * 4);
  unsigned short* h2B  = (unsigned short*)alloc((size_t)MS * D_ * 2);
  unsigned short* ff1  = (unsigned short*)alloc((size_t)MS * DFF_ * 2);

  // 1. weight convert + transpose to bf16 [N,K]
  convt_kernel<<<(D_ * D_  + 255) / 256, 256, 0, stream>>>(Wq, WqT, D_, D_);
  convt_kernel<<<(D_ * D_  + 255) / 256, 256, 0, stream>>>(Wk, WkT, D_, D_);
  convt_kernel<<<(D_ * D_  + 255) / 256, 256, 0, stream>>>(Wv, WvT, D_, D_);
  convt_kernel<<<(D_ * D_  + 255) / 256, 256, 0, stream>>>(Wo, WoT, D_, D_);
  convt_kernel<<<(D_ * DFF_ + 255) / 256, 256, 0, stream>>>(W1, W1T, D_, DFF_);
  convt_kernel<<<(D_ * DFF_ + 255) / 256, 256, 0, stream>>>(W2, W2T, DFF_, D_);

  // 2. LN1 -> bf16
  ln_bf16_kernel<<<MS, 256, 0, stream>>>(x, ln1_g, ln1_b, hB);

  // 3. QKV projections (WMMA)
  dim3 gD(D_ / 128, MS / 128), gF(DFF_ / 128, MS / 128);
  gemm_bf16_kernel<<<gD, 256, 0, stream>>>(hB, WqT, bq, nullptr, nullptr, Qh, MS, D_, D_, S, MODE_QK);
  gemm_bf16_kernel<<<gD, 256, 0, stream>>>(hB, WkT, bk, nullptr, nullptr, Kh, MS, D_, D_, S, MODE_QK);
  gemm_bf16_kernel<<<gD, 256, 0, stream>>>(hB, WvT, bv, nullptr, nullptr, Vt, MS, D_, D_, S, MODE_VT);

  // 4. flash attention (WMMA)
  attn_kernel<<<dim3(S / 16, B * H_), 32, 0, stream>>>(Qh, Kh, Vt, amask, ctxB, S);

  // 5. output projection + residual -> x2 (f32)
  gemm_bf16_kernel<<<gD, 256, 0, stream>>>(ctxB, WoT, bo, x, x2, nullptr, MS, D_, D_, S, MODE_F32_RES);

  // 6. LN2 -> bf16
  ln_bf16_kernel<<<MS, 256, 0, stream>>>(x2, ln2_g, ln2_b, h2B);

  // 7. FFN1: relu(h2 @ W1 + b1) -> bf16
  gemm_bf16_kernel<<<gF, 256, 0, stream>>>(h2B, W1T, b1, nullptr, nullptr, ff1, MS, DFF_, D_, S, MODE_BF16_RELU);

  // 8. FFN2 + residual -> d_out (f32)
  gemm_bf16_kernel<<<gD, 256, 0, stream>>>(ff1, W2T, b2, x2, (float*)d_out, nullptr, MS, D_, DFF_, S, MODE_F32_RES);
}
